// MultihAttention_8409545966294
// MI455X (gfx1250) — matrix operand order repaired and checked by reading, not compiled, not run on hardware
//
#include <hip/hip_runtime.h>

// ---------------------------------------------------------------------------
// Linear multi-head attention, MI455X (gfx1250), wave32 + WMMA bf16.
//   1. prep_weights : W* f32 row-major -> bf16 K-contiguous (transposed)
//   2. gemm512_wmma : Q,K (f32), V (bf16) projections
//                     (async global->LDS B staging + pipelined WMMA)
//   3. q_softmax    : softmax over d_k=64 per row, *1/8, -> bf16
//   4. k_colmax/k_expsum : seq-dim max, exp(K-max) -> bf16, column sums
//   5. context_kernel : ctx[b,h] = exp_kT @ V  (64x64/head, f32 atomics)
//   6. ctx_norm     : ctx /= colsum, transpose to K-contiguous bf16
//   7. attn_ctx_gemm: attn = q_soft @ ctx (WMMA, K=64)
//   8. gemm512_wmma : out = attn @ Wo + bo (f32 output)
// ---------------------------------------------------------------------------

typedef __attribute__((ext_vector_type(16))) __bf16  v16bf;
typedef __attribute__((ext_vector_type(8)))  __bf16  bf16x8;
typedef __attribute__((ext_vector_type(8)))  float   v8f;
typedef __attribute__((ext_vector_type(4)))  float   f32x4;

union V16U { v16bf v; bf16x8 h[2]; };

// Async global->LDS copies via inline asm (CDNA5 ISA §15.18.3 / §9 SOPP).
// Flip to 0 to fall back to plain load + ds_store staging.
#define ASYNC_LDS 1

__device__ __forceinline__ void wait_async_done() {
#if ASYNC_LDS
  asm volatile("s_wait_asynccnt 0x0" ::: "memory");
#endif
}

// Copy 16 B per lane from global to LDS (async, tracked by ASYNCcnt).
__device__ __forceinline__ void async_copy16(const __bf16* src, __bf16* dst_lds) {
#if ASYNC_LDS
  // Generic LDS address: low 32 bits are the LDS byte offset (ISA 10.2).
  unsigned loff = (unsigned)(unsigned long long)(uintptr_t)dst_lds;
  unsigned long long ga = (unsigned long long)(uintptr_t)src;
  asm volatile("global_load_async_to_lds_b128 %0, %1, off"
               :: "v"(loff), "v"(ga) : "memory");
#else
  *reinterpret_cast<bf16x8*>(dst_lds) = *reinterpret_cast<const bf16x8*>(src);
#endif
}

// Copy one 64-col x 32-K bf16 B slice (4 KB) into LDS; 128 threads x 2 x 16B.
__device__ __forceinline__ void stage_b_slice(const __bf16* __restrict__ Bt,
                                              int n0, int k0, __bf16* dst) {
  const int tid = threadIdx.x;
#pragma unroll
  for (int i = 0; i < 2; ++i) {
    int q   = tid + (i << 7);       // 256 chunks of 16 B
    int col = q >> 2;
    int kq  = (q & 3) << 3;
    const __bf16* src = Bt + (size_t)(n0 + col) * 512 + k0 + kq;
    async_copy16(src, dst + col * 32 + kq);
  }
}

// A fragment (16x32, MxK): lane L holds row M=L%16, K = lhi*8 + {0..7} and +16.
__device__ __forceinline__ v16bf load_a_frag(const float* rowp, int kbase) {
  const f32x4* p0 = reinterpret_cast<const f32x4*>(rowp + kbase);
  const f32x4* p1 = reinterpret_cast<const f32x4*>(rowp + kbase + 16);
  f32x4 a0 = p0[0], a1 = p0[1], a2 = p1[0], a3 = p1[1];
  V16U u;
#pragma unroll
  for (int i = 0; i < 4; ++i) {
    u.v[i]      = (__bf16)a0[i];
    u.v[4 + i]  = (__bf16)a1[i];
    u.v[8 + i]  = (__bf16)a2[i];
    u.v[12 + i] = (__bf16)a3[i];
  }
  return u.v;
}

__device__ __forceinline__ v16bf load_a_frag(const __bf16* rowp, int kbase) {
  V16U u;
  u.h[0] = *reinterpret_cast<const bf16x8*>(rowp + kbase);
  u.h[1] = *reinterpret_cast<const bf16x8*>(rowp + kbase + 16);
  return u.v;
}

// B fragment (32x16, KxN): lane L holds col N=L%16, K = lhi*8 + {0..7} and +16.
__device__ __forceinline__ v16bf load_b_frag(const __bf16* colp, int kbase) {
  V16U u;
  u.h[0] = *reinterpret_cast<const bf16x8*>(colp + kbase);
  u.h[1] = *reinterpret_cast<const bf16x8*>(colp + kbase + 16);
  return u.v;
}

// B fragment from an LDS slice laid out [col][32 K].
__device__ __forceinline__ v16bf lds_b_frag(const __bf16* base, int c, int klo) {
  V16U u;
  u.h[0] = *reinterpret_cast<const bf16x8*>(base + c * 32 + klo);
  u.h[1] = *reinterpret_cast<const bf16x8*>(base + c * 32 + klo + 16);
  return u.v;
}

__device__ __forceinline__ v8f wmma_bf16(v16bf a, v16bf b, v8f c) {
  return __builtin_amdgcn_wmma_f32_16x16x32_bf16(
      false, a, false, b, (short)0, c, false, false);
}

// ------------------------------ GEMM (K=N=512) ------------------------------
// block = 128 threads = 4 waves; wave -> 32x64 of C; block -> 128x64.
// B strip staged once per block through LDS (async copies, double buffered);
// A stream register-double-buffered from global.
template <typename AT, bool OUT_BF16>
__global__ __launch_bounds__(128) void gemm512_wmma(
    const AT* __restrict__ A, const __bf16* __restrict__ Bt,
    const float* __restrict__ bias, void* __restrict__ Cout) {
  __shared__ __bf16 sB[2][64 * 32];
  const int wave = threadIdx.x >> 5;
  const int lane = threadIdx.x & 31;
  const int lrow = lane & 15;
  const int lhi  = lane >> 4;
  const int m0 = blockIdx.x * 128 + wave * 32;
  const int n0 = blockIdx.y * 64;
  const int ka = lhi * 8;
  const int kb = lhi * 8;

  const AT* Arow0 = A + (size_t)(m0 + lrow) * 512;
  const AT* Arow1 = Arow0 + 16 * 512;

  v8f acc[2][4] = {};
  stage_b_slice(Bt, n0, 0, sB[0]);
  v16bf a0 = load_a_frag(Arow0, ka);
  v16bf a1 = load_a_frag(Arow1, ka);

  for (int ks = 0; ks < 16; ++ks) {
    wait_async_done();       // own wave's slice-ks copies complete
    __syncthreads();         // all waves' copies landed; prev compute done
    if (ks < 15)
      stage_b_slice(Bt, n0, (ks + 1) * 32, sB[(ks + 1) & 1]);
    v16bf na0 = a0, na1 = a1;
    if (ks < 15) {
      na0 = load_a_frag(Arow0, (ks + 1) * 32 + ka);
      na1 = load_a_frag(Arow1, (ks + 1) * 32 + ka);
      __builtin_prefetch(Arow0 + (ks + 2) * 32, 0, 3);  // global_prefetch_b8
      __builtin_prefetch(Arow1 + (ks + 2) * 32, 0, 3);
    }
    const __bf16* bufp = sB[ks & 1];
#pragma unroll
    for (int t = 0; t < 4; ++t) {
      v16bf bb = lds_b_frag(bufp, t * 16 + lrow, kb);
      acc[0][t] = wmma_bf16(a0, bb, acc[0][t]);
      acc[1][t] = wmma_bf16(a1, bb, acc[1][t]);
    }
    a0 = na0; a1 = na1;
  }

#pragma unroll
  for (int mt = 0; mt < 2; ++mt) {
#pragma unroll
    for (int t = 0; t < 4; ++t) {
      const int col = n0 + t * 16 + lrow;
      const float bvv = bias[col];
#pragma unroll
      for (int j = 0; j < 8; ++j) {
        const int row = m0 + mt * 16 + j + lhi * 8;  // lanes 16-31 -> M=8+j
        float vv = acc[mt][t][j] + bvv;
        if (OUT_BF16)
          reinterpret_cast<__bf16*>(Cout)[(size_t)row * 512 + col] = (__bf16)vv;
        else
          reinterpret_cast<float*>(Cout)[(size_t)row * 512 + col] = vv;
      }
    }
  }
}

// --------------------- weight transpose + bf16 convert ----------------------
__global__ __launch_bounds__(256) void prep_weights(
    const float* __restrict__ Wq, const float* __restrict__ Wk,
    const float* __restrict__ Wv, const float* __restrict__ Wo,
    __bf16* __restrict__ Wt) {
  int idx = blockIdx.x * 256 + threadIdx.x;        // 4 * 512 * 512
  int w   = idx >> 18;
  int rem = idx & 262143;
  int n = rem >> 9, k = rem & 511;                 // output: [n][k]
  const float* W = (w == 0) ? Wq : (w == 1) ? Wk : (w == 2) ? Wv : Wo;
  Wt[idx] = (__bf16)W[k * 512 + n];                // input: [k][n]
}

// ------------------------------- init stats ---------------------------------
__global__ __launch_bounds__(256) void init_stats(
    float* __restrict__ kmax, float* __restrict__ ksum, float* __restrict__ ctx) {
  int i = blockIdx.x * 256 + threadIdx.x;          // 65536 threads
  if (i < 1024) { kmax[i] = -3.0e38f; ksum[i] = 0.0f; }
  ctx[i] = 0.0f;
}

// ---------------------- q softmax over d_k (64), scaled ---------------------
__global__ __launch_bounds__(256) void q_softmax_kernel(
    const float* __restrict__ Q, __bf16* __restrict__ qs) {
  const int wave = threadIdx.x >> 5;
  const int lane = threadIdx.x & 31;
  const int seg  = blockIdx.x * 8 + wave;          // (row, head) segment of 64
  const float* base = Q + (size_t)seg * 64;
  float2 x = *reinterpret_cast<const float2*>(base + lane * 2);
  float m = fmaxf(x.x, x.y);
#pragma unroll
  for (int off = 16; off >= 1; off >>= 1) m = fmaxf(m, __shfl_xor(m, off));
  float e0 = __expf(x.x - m), e1 = __expf(x.y - m);
  float s = e0 + e1;
#pragma unroll
  for (int off = 16; off >= 1; off >>= 1) s += __shfl_xor(s, off);
  float inv = 0.125f / s;                          // 1/sqrt(d_k=64) folded in
  __bf16* o = qs + (size_t)seg * 64 + lane * 2;
  o[0] = (__bf16)(e0 * inv);
  o[1] = (__bf16)(e1 * inv);
}

// ----------------------- k column stats over sequence -----------------------
__device__ __forceinline__ void atomicMaxFloat(float* addr, float val) {
  if (val >= 0.0f)
    atomicMax(reinterpret_cast<int*>(addr), __float_as_int(val));
  else
    atomicMin(reinterpret_cast<unsigned int*>(addr), __float_as_uint(val));
}

__global__ __launch_bounds__(256) void k_colmax(
    const float* __restrict__ K, float* __restrict__ kmax) {
  const int chunk = blockIdx.x;                    // 256 chunks of 128 rows
  const int b  = chunk >> 7;
  const int n0 = (chunk & 127) * 128;
  const int c0 = threadIdx.x * 2;
  const float* base = K + ((size_t)b * 16384 + n0) * 512 + c0;
  float m0 = -3.0e38f, m1 = -3.0e38f;
  for (int r = 0; r < 128; ++r) {
    float2 x = *reinterpret_cast<const float2*>(base + (size_t)r * 512);
    m0 = fmaxf(m0, x.x); m1 = fmaxf(m1, x.y);
  }
  atomicMaxFloat(kmax + b * 512 + c0, m0);
  atomicMaxFloat(kmax + b * 512 + c0 + 1, m1);
}

__global__ __launch_bounds__(256) void k_expsum(
    const float* __restrict__ K, const float* __restrict__ kmax,
    __bf16* __restrict__ ek, float* __restrict__ ksum) {
  const int chunk = blockIdx.x;
  const int b  = chunk >> 7;
  const int n0 = (chunk & 127) * 128;
  const int c0 = threadIdx.x * 2;
  const float mx0 = kmax[b * 512 + c0];
  const float mx1 = kmax[b * 512 + c0 + 1];
  const float* base = K + ((size_t)b * 16384 + n0) * 512 + c0;
  __bf16* obase = ek + ((size_t)b * 16384 + n0) * 512 + c0;
  float s0 = 0.0f, s1 = 0.0f;
  for (int r = 0; r < 128; ++r) {
    float2 x = *reinterpret_cast<const float2*>(base + (size_t)r * 512);
    float e0 = __expf(x.x - mx0), e1 = __expf(x.y - mx1);
    obase[(size_t)r * 512]     = (__bf16)e0;
    obase[(size_t)r * 512 + 1] = (__bf16)e1;
    s0 += e0; s1 += e1;
  }
  atomicAdd(ksum + b * 512 + c0, s0);
  atomicAdd(ksum + b * 512 + c0 + 1, s1);
}

// -------------------- ctx[b,h] = exp_k^T @ V (64x64/head) -------------------
// 8 rows staged in LDS per barrier; each thread owns a 4x4 output block.
__global__ __launch_bounds__(256) void context_kernel(
    const __bf16* __restrict__ ek, const __bf16* __restrict__ v,
    float* __restrict__ ctx) {
  __shared__ float se[8][64], sv[8][64];
  const int bh = blockIdx.x;                       // 16
  const int b = bh >> 3, h = bh & 7;
  const size_t rowbase = ((size_t)b * 16384 + blockIdx.y * 1024) * 512 + h * 64;
  const int t  = threadIdx.x;
  const int d0 = (t >> 4) * 4, e0 = (t & 15) * 4;
  float acc[4][4] = {};
  for (int r0 = 0; r0 < 1024; r0 += 8) {
#pragma unroll
    for (int i = 0; i < 4; ++i) {
      int idx = t + i * 256;                       // 1024 = 2 mats * 8 rows * 64
      int mat = idx >> 9, row = (idx >> 6) & 7, col = idx & 63;
      size_t off = rowbase + (size_t)(r0 + row) * 512 + col;
      if (mat) sv[row][col] = (float)v[off];
      else     se[row][col] = (float)ek[off];
    }
    __syncthreads();
#pragma unroll
    for (int rr = 0; rr < 8; ++rr) {
      float ev[4], vv[4];
#pragma unroll
      for (int i = 0; i < 4; ++i) { ev[i] = se[rr][d0 + i]; vv[i] = sv[rr][e0 + i]; }
#pragma unroll
      for (int i = 0; i < 4; ++i)
#pragma unroll
        for (int j = 0; j < 4; ++j) acc[i][j] += ev[i] * vv[j];
    }
    __syncthreads();
  }
  float* cbase = ctx + (size_t)bh * 4096;
#pragma unroll
  for (int i = 0; i < 4; ++i)
#pragma unroll
    for (int j = 0; j < 4; ++j)
      atomicAdd(cbase + (d0 + i) * 64 + (e0 + j), acc[i][j]);
}

// ------------- normalize ctx by column sums, transpose, -> bf16 -------------
__global__ __launch_bounds__(256) void ctx_norm(
    const float* __restrict__ ctx, const float* __restrict__ ksum,
    __bf16* __restrict__ ctxT) {
  int idx = blockIdx.x * 256 + threadIdx.x;        // 65536
  int bh = idx >> 12, d = (idx >> 6) & 63, e = idx & 63;
  int b = bh >> 3, h = bh & 7;
  float s = ksum[b * 512 + h * 64 + d];
  ctxT[(size_t)bh * 4096 + e * 64 + d] = (__bf16)(ctx[idx] / s);
}

// ------------------ attn = q_soft @ ctx (per-head GEMM K=64) ----------------
__global__ __launch_bounds__(128) void attn_ctx_gemm(
    const __bf16* __restrict__ qsoft, const __bf16* __restrict__ ctxT,
    __bf16* __restrict__ attn) {
  const int wave = threadIdx.x >> 5;
  const int lane = threadIdx.x & 31;
  const int lrow = lane & 15;
  const int lhi  = lane >> 4;
  const int h  = blockIdx.y;
  const int m0 = blockIdx.x * 128 + wave * 32;
  const int b  = m0 >> 14;
  const int ka = lhi * 8, kb = lhi * 8;
  const __bf16* ctx   = ctxT + (size_t)(b * 8 + h) * 4096;
  const __bf16* Arow0 = qsoft + (size_t)(m0 + lrow) * 512 + h * 64;
  const __bf16* Arow1 = Arow0 + 16 * 512;

  v8f acc[2][4] = {};
#pragma unroll
  for (int k0 = 0; k0 < 64; k0 += 32) {
    v16bf a0 = load_a_frag(Arow0, k0 + ka);
    v16bf a1 = load_a_frag(Arow1, k0 + ka);
#pragma unroll
    for (int t = 0; t < 4; ++t) {
      v16bf bb = load_b_frag(ctx + (t * 16 + lrow) * 64, k0 + kb);
      acc[0][t] = wmma_bf16(a0, bb, acc[0][t]);
      acc[1][t] = wmma_bf16(a1, bb, acc[1][t]);
    }
  }
#pragma unroll
  for (int mt = 0; mt < 2; ++mt) {
#pragma unroll
    for (int t = 0; t < 4; ++t) {
      const int col = h * 64 + t * 16 + lrow;
#pragma unroll
      for (int j = 0; j < 8; ++j) {
        const int row = m0 + mt * 16 + j + lhi * 8;
        attn[(size_t)row * 512 + col] = (__bf16)acc[mt][t][j];
      }
    }
  }
}

// ----------------------------------------------------------------------------
extern "C" void kernel_launch(void* const* d_in, const int* in_sizes, int n_in,
                              void* d_out, int out_size, void* d_ws, size_t ws_size,
                              hipStream_t stream) {
  (void)in_sizes; (void)n_in; (void)out_size; (void)ws_size;
  const float* query = (const float*)d_in[0];
  const float* key   = (const float*)d_in[1];
  const float* value = (const float*)d_in[2];
  const float* Wq = (const float*)d_in[3];
  const float* bq = (const float*)d_in[4];
  const float* Wk = (const float*)d_in[5];
  const float* bk = (const float*)d_in[6];
  const float* Wv = (const float*)d_in[7];
  const float* bv = (const float*)d_in[8];
  const float* Wo = (const float*)d_in[9];
  const float* bo = (const float*)d_in[10];
  float* out = (float*)d_out;

  char* p = (char*)d_ws;
  auto take = [&](size_t bytes) -> char* {
    char* r = p; p += (bytes + 255) & ~(size_t)255; return r;
  };
  __bf16* Wt   = (__bf16*)take(4ull * 512 * 512 * 2);   //  2 MiB transposed weights
  float*  Qf   = (float*) take(32768ull * 512 * 4);     // 64 MiB Q projection
  float*  Kf   = (float*) take(32768ull * 512 * 4);     // 64 MiB K projection
  __bf16* Vb   = (__bf16*)take(32768ull * 512 * 2);     // 32 MiB V projection (bf16)
  __bf16* Qs   = (__bf16*)take(32768ull * 512 * 2);     // 32 MiB softmaxed q
  __bf16* Ek   = (__bf16*)take(32768ull * 512 * 2);     // 32 MiB exp(k - max)
  __bf16* At   = (__bf16*)take(32768ull * 512 * 2);     // 32 MiB attention out
  float*  Km   = (float*) take(1024 * 4);               // column max
  float*  Ks   = (float*) take(1024 * 4);               // column sum(exp)
  float*  Ctx  = (float*) take(65536ull * 4);           // [bh][64][64]
  __bf16* CtxT = (__bf16*)take(65536ull * 2);           // normalized, transposed

  const __bf16* Wq_t = Wt;
  const __bf16* Wk_t = Wt + 262144;
  const __bf16* Wv_t = Wt + 2 * 262144;
  const __bf16* Wo_t = Wt + 3 * 262144;

  init_stats<<<256, 256, 0, stream>>>(Km, Ks, Ctx);
  prep_weights<<<4096, 256, 0, stream>>>(Wq, Wk, Wv, Wo, Wt);

  gemm512_wmma<float, false><<<dim3(256, 8), 128, 0, stream>>>(query, Wq_t, bq, Qf);
  gemm512_wmma<float, false><<<dim3(256, 8), 128, 0, stream>>>(key,   Wk_t, bk, Kf);
  gemm512_wmma<float, true ><<<dim3(256, 8), 128, 0, stream>>>(value, Wv_t, bv, Vb);

  q_softmax_kernel<<<32768, 256, 0, stream>>>(Qf, Qs);
  k_colmax<<<256, 256, 0, stream>>>(Kf, Km);
  k_expsum<<<256, 256, 0, stream>>>(Kf, Km, Ek, Ks);

  context_kernel<<<dim3(16, 16), 256, 0, stream>>>(Ek, Vb, Ctx);
  ctx_norm<<<256, 256, 0, stream>>>(Ctx, Ks, CtxT);

  attn_ctx_gemm<<<dim3(256, 8), 128, 0, stream>>>(Qs, CtxT, At);
  gemm512_wmma<__bf16, false><<<dim3(256, 8), 128, 0, stream>>>(At, Wo_t, bo, out);
}
